// MultiScaleSynergisticFusion_79620103733825
// MI455X (gfx1250) — compile-verified, matrix operand order checked by reading, not run on hardware
//
#include <hip/hip_runtime.h>
#include <hip/hip_bf16.h>

#define BATCH 4
#define NPIX  4096      // H*W = 64*64
#define CCH   256
#define DD    32        // C/8
#define KCH   64        // keys per chunk

typedef __attribute__((ext_vector_type(16))) __bf16 v16bf;
typedef __attribute__((ext_vector_type(8)))  float  v8f;

union Frag16 { uint4 u[2]; v16bf v; };

__device__ __forceinline__ unsigned short f2bf(float f) {
  union { float f; unsigned u; } x; x.f = f;
  unsigned r = x.u + 0x7FFFu + ((x.u >> 16) & 1u);   // round-to-nearest-even
  return (unsigned short)(r >> 16);
}
__device__ __forceinline__ unsigned pack2bf(float a, float b) {
  return (unsigned)f2bf(a) | ((unsigned)f2bf(b) << 16);
}

// ---------------- Projection: Q and K (one thread per pixel) ----------------
__global__ __launch_bounds__(256)
void proj_qk(const float* __restrict__ sem, const float* __restrict__ str,
             const float* __restrict__ Wq,  const float* __restrict__ bq,
             const float* __restrict__ Wk,  const float* __restrict__ bk,
             unsigned short* __restrict__ Qb, unsigned short* __restrict__ Kb) {
  int idx = blockIdx.x * blockDim.x + threadIdx.x;   // B*N threads
  int b = idx >> 12;
  int i = idx & (NPIX - 1);
  float qa[DD], ka[DD];
#pragma unroll
  for (int d = 0; d < DD; ++d) { qa[d] = bq[d]; ka[d] = bk[d]; }
  const float* sp = sem + (size_t)b * CCH * NPIX + i;
  const float* tp = str + (size_t)b * CCH * NPIX + i;
  for (int c = 0; c < CCH; ++c) {
    float s = sp[(size_t)c * NPIX];
    float t = tp[(size_t)c * NPIX];
#pragma unroll
    for (int d = 0; d < DD; ++d) {
      qa[d] = __builtin_fmaf(Wq[d * CCH + c], s, qa[d]);
      ka[d] = __builtin_fmaf(Wk[d * CCH + c], t, ka[d]);
    }
  }
  unsigned short* qo = Qb + (size_t)idx * DD;   // layout [b][i][d], 64B rows
  unsigned short* ko = Kb + (size_t)idx * DD;   // layout [b][j][d]
#pragma unroll
  for (int d = 0; d < DD; ++d) { qo[d] = f2bf(qa[d]); ko[d] = f2bf(ka[d]); }
}

// ---------------- Projection: V (one thread per (b, 8 channels, pixel)) ----
__global__ __launch_bounds__(256)
void proj_v(const float* __restrict__ str, const float* __restrict__ Wv,
            const float* __restrict__ bv,  unsigned short* __restrict__ Vb) {
  int idx = blockIdx.x * blockDim.x + threadIdx.x;   // B*32*N threads
  int j  = idx & (NPIX - 1);
  int t  = idx >> 12;          // b*32 + c8
  int b  = t >> 5;
  int c8 = t & 31;
  float va[8];
#pragma unroll
  for (int k = 0; k < 8; ++k) va[k] = bv[c8 * 8 + k];
  const float* tp = str + (size_t)b * CCH * NPIX + j;
  for (int cc = 0; cc < CCH; ++cc) {
    float s = tp[(size_t)cc * NPIX];
#pragma unroll
    for (int k = 0; k < 8; ++k)
      va[k] = __builtin_fmaf(Wv[(c8 * 8 + k) * CCH + cc], s, va[k]);
  }
#pragma unroll
  for (int k = 0; k < 8; ++k)                       // layout [b][c][j]
    Vb[((size_t)(b * CCH + c8 * 8 + k)) * NPIX + j] = f2bf(va[k]);
}

// ---------------- Fused flash attention + gamma residual ----------------
// Block = 128 threads = 4 waves; each wave owns 16 query rows.
// Scores computed TRANSPOSED (S^T = K*Q^T) so each lane owns one query's row
// of scores in registers -> softmax reductions are in-lane + one xor16.
// 64-key chunks amortize barriers / acc-rescale over 36 WMMAs per iteration.
__global__ __launch_bounds__(128)
void attn_fused(const unsigned short* __restrict__ Qb,
                const unsigned short* __restrict__ Kb,
                const unsigned short* __restrict__ Vb,
                const float* __restrict__ sem,
                const float* __restrict__ gamma,
                float* __restrict__ out) {
  __shared__ __align__(16) unsigned short lds_v[2][CCH * KCH];  // double-buffered V chunk [c][j], 2x32KB
  __shared__ __align__(16) unsigned short lds_p[4][16 * KCH];   // per-wave P tile [query][key], 8KB
  __shared__ __align__(16) float          lds_s[4][16];         // per-wave per-query broadcast

  const int lane = threadIdx.x & 31;
  const int wave = threadIdx.x >> 5;
  const int n16  = lane & 15;
  const int h    = lane >> 4;

  const int nqb   = NPIX / 64;
  const int b     = blockIdx.x / nqb;
  const int qbase = (blockIdx.x % nqb) * 64 + wave * 16;

  // Q fragment (used as B-matrix: 32(D) x 16(query)): col = query = n16,
  // element runs D=[8h..8h+7] and [16+8h..23+8h]  -> two 16B loads
  Frag16 qf;
  {
    const uint4* qp = (const uint4*)(Qb + ((size_t)(b * NPIX + qbase + n16)) * DD);
    qf.u[0] = qp[h];
    qf.u[1] = qp[2 + h];
  }

  // async-DMA one 64-key V chunk (all 256 channels) into LDS buffer `buf`
  auto issueV = [&](int j0, int buf) {
#pragma unroll
    for (int rr = 0; rr < 2; ++rr) {
      int c = threadIdx.x + rr * 128;
      unsigned long long g =
          (unsigned long long)(Vb + ((size_t)(b * CCH + c)) * NPIX + j0);
      unsigned l = (unsigned)(unsigned long long)(&lds_v[buf][c * KCH]);
#pragma unroll
      for (int q4 = 0; q4 < 8; ++q4) {       // 128B per V row
        asm volatile("global_load_async_to_lds_b128 %0, %1, off"
                     :: "v"(l + q4 * 16), "v"(g + q4 * 16) : "memory");
      }
    }
  };

  float mq = -1e30f, lq = 0.f;      // per-lane softmax state for query n16
  v8f acc[16];
#pragma unroll
  for (int t = 0; t < 16; ++t)
#pragma unroll
    for (int r = 0; r < 8; ++r) acc[t][r] = 0.f;

  const float LOG2E = 1.44269504f;

  issueV(0, 0);

  for (int j0 = 0; j0 < NPIX; j0 += KCH) {
    const int p = (j0 >> 6) & 1;
    asm volatile("s_wait_asynccnt 0x0" ::: "memory");   // chunk p has landed
    __syncthreads();                                    // ...for every wave
    if (j0 + KCH < NPIX) issueV(j0 + KCH, 1 - p);       // prefetch next chunk

    // ---- transposed scores: S^T tiles = K_tile(A) * Q(B), 4 key tiles ----
    v8f st[4];
    v8f z;
#pragma unroll
    for (int r = 0; r < 8; ++r) z[r] = 0.f;
#pragma unroll
    for (int jt = 0; jt < 4; ++jt) {
      Frag16 kf;
      const uint4* kp =
          (const uint4*)(Kb + ((size_t)(b * NPIX + j0 + 16 * jt + n16)) * DD);
      kf.u[0] = kp[h]; kf.u[1] = kp[2 + h];
      // D-frag: M = key (16*jt + r + 8h within chunk), N = query (n16)
      st[jt] = __builtin_amdgcn_wmma_f32_16x16x32_bf16(false, kf.v, false, qf.v,
                                                       (short)0, z, false, false);
    }

    // ---- online softmax: lane owns query n16 ----
    float rm = st[0][0];
#pragma unroll
    for (int jt = 0; jt < 4; ++jt)
#pragma unroll
      for (int r = 0; r < 8; ++r) rm = fmaxf(rm, st[jt][r]);
    rm = fmaxf(rm, __shfl_xor(rm, 16));                 // combine half-waves
    float mn   = fmaxf(mq, rm);
    float corr = exp2f((mq - mn) * LOG2E);
    float rs = 0.f;
#pragma unroll
    for (int jt = 0; jt < 4; ++jt) {
      float e[8];
#pragma unroll
      for (int r = 0; r < 8; ++r) { e[r] = exp2f((st[jt][r] - mn) * LOG2E); rs += e[r]; }
      uint4 w = { pack2bf(e[0], e[1]), pack2bf(e[2], e[3]),
                  pack2bf(e[4], e[5]), pack2bf(e[6], e[7]) };
      // keys 16*jt + 8h .. +7 of query-row n16
      *(uint4*)&lds_p[wave][n16 * KCH + 16 * jt + 8 * h] = w;
    }
    rs += __shfl_xor(rs, 16);
    lq = lq * corr + rs;
    mq = mn;
    if (h == 0) lds_s[wave][n16] = corr;                // broadcast scale

    // same-wave LDS RAW hazard: stores above feed loads below
    asm volatile("s_wait_dscnt 0x0" ::: "memory");

    // rescale accumulators: acc row r holds query (r + 8h)
    float scl[8];
    {
      const float4* s4 = (const float4*)&lds_s[wave][8 * h];
      float4 sa = s4[0], sb = s4[1];
      scl[0] = sa.x; scl[1] = sa.y; scl[2] = sa.z; scl[3] = sa.w;
      scl[4] = sb.x; scl[5] = sb.y; scl[6] = sb.z; scl[7] = sb.w;
    }
#pragma unroll
    for (int t = 0; t < 16; ++t)
#pragma unroll
      for (int r = 0; r < 8; ++r) acc[t][r] *= scl[r];

    // ---- PV: two P A-frags (16q x 32k each) x V B-frags (32k x 16ch) ----
    Frag16 pf0, pf1;
    {
      const uint4* pp = (const uint4*)(&lds_p[wave][n16 * KCH]);  // 128B row
      pf0.u[0] = pp[h];     pf0.u[1] = pp[2 + h];                 // keys 0..31
      pf1.u[0] = pp[4 + h]; pf1.u[1] = pp[6 + h];                 // keys 32..63
    }
    const unsigned short* vbuf = lds_v[p];
#pragma unroll
    for (int t = 0; t < 16; ++t) {
      const uint4* vp = (const uint4*)(vbuf + (t * 16 + n16) * KCH);
      Frag16 vf;
      vf.u[0] = vp[h]; vf.u[1] = vp[2 + h];                       // keys 0..31
      acc[t] = __builtin_amdgcn_wmma_f32_16x16x32_bf16(false, pf0.v, false, vf.v,
                                                       (short)0, acc[t], false, false);
      vf.u[0] = vp[4 + h]; vf.u[1] = vp[6 + h];                   // keys 32..63
      acc[t] = __builtin_amdgcn_wmma_f32_16x16x32_bf16(false, pf1.v, false, vf.v,
                                                       (short)0, acc[t], false, false);
    }
    __syncthreads();   // everyone done reading buffer p before it is re-filled
  }

  // ---- epilogue: out = gamma * (acc / l) + semantic ----
  if (h == 0) lds_s[wave][n16] = lq;
  asm volatile("s_wait_dscnt 0x0" ::: "memory");
  float linv[8];
  {
    const float4* s4 = (const float4*)&lds_s[wave][8 * h];
    float4 sa = s4[0], sb = s4[1];
    linv[0] = 1.f / sa.x; linv[1] = 1.f / sa.y; linv[2] = 1.f / sa.z; linv[3] = 1.f / sa.w;
    linv[4] = 1.f / sb.x; linv[5] = 1.f / sb.y; linv[6] = 1.f / sb.z; linv[7] = 1.f / sb.w;
  }
  const float g = gamma[0];
#pragma unroll
  for (int t = 0; t < 16; ++t) {
#pragma unroll
    for (int r = 0; r < 8; ++r) {
      int c = t * 16 + n16;
      int i = qbase + r + 8 * h;
      size_t idx = ((size_t)(b * CCH + c)) * NPIX + i;
      out[idx] = g * (acc[t][r] * linv[r]) + sem[idx];
    }
  }
}

extern "C" void kernel_launch(void* const* d_in, const int* in_sizes, int n_in,
                              void* d_out, int out_size, void* d_ws, size_t ws_size,
                              hipStream_t stream) {
  const float* sem   = (const float*)d_in[0];
  const float* str   = (const float*)d_in[1];
  const float* Wq    = (const float*)d_in[2];
  const float* bq    = (const float*)d_in[3];
  const float* Wk    = (const float*)d_in[4];
  const float* bk    = (const float*)d_in[5];
  const float* Wv    = (const float*)d_in[6];
  const float* bv    = (const float*)d_in[7];
  const float* gamma = (const float*)d_in[8];
  float* out = (float*)d_out;

  // workspace: Q (1MB) | K (1MB) | V (8MB), all bf16
  unsigned short* Qb = (unsigned short*)d_ws;
  unsigned short* Kb = Qb + (size_t)BATCH * NPIX * DD;
  unsigned short* Vb = Kb + (size_t)BATCH * NPIX * DD;

  proj_qk<<<(BATCH * NPIX) / 256, 256, 0, stream>>>(sem, str, Wq, bq, Wk, bk, Qb, Kb);
  proj_v<<<(BATCH * 32 * NPIX) / 256, 256, 0, stream>>>(str, Wv, bv, Vb);
  attn_fused<<<BATCH * (NPIX / 64), 128, 0, stream>>>(Qb, Kb, Vb, sem, gamma, out);
}
